// MLPBlock_27736898798389
// MI455X (gfx1250) — compile-verified, matrix operand order checked by reading, not run on hardware
//
#include <hip/hip_runtime.h>

typedef __bf16 bf16;
typedef __attribute__((ext_vector_type(16))) __bf16 v16bf;
typedef __attribute__((ext_vector_type(8)))  __bf16 v8bf;
typedef __attribute__((ext_vector_type(8)))  float  v8f;
typedef __attribute__((ext_vector_type(4))) unsigned int u32x4;
typedef __attribute__((ext_vector_type(8))) int          i32x8;
typedef __attribute__((ext_vector_type(4))) int          i32x4;

#define T_TOK 2048
#define HDIM  1024
#define IDIM  1024
#define I2    2048
#define NEXP  16
#define TOPK  4
#define ALPHA 1.702f
#define LIMIT 7.0f
#define AS_STR 1032    // gemm1 LDS row stride (bf16 elems)
#define HR_B  1040     // gemm2 LDS half-row pitch: 1024B data + 16B pad (TDM pad cfg)

#if __has_builtin(__builtin_amdgcn_tensor_load_to_lds)
#define HAS_TDM 1
#else
#define HAS_TDM 0
#endif

// ---- workspace layout (bytes) ----
static constexpr size_t OFF_CNT  = 0;                            // 16 ints
static constexpr size_t OFF_BASE = 256;                          // 16 ints
static constexpr size_t OFF_TOK  = 512;                          // E*T ints  = 128KB
static constexpr size_t OFF_WGT  = OFF_TOK + (size_t)NEXP*T_TOK*4;        // E*T floats
static constexpr size_t OFF_TBF  = ((OFF_WGT + (size_t)NEXP*T_TOK*4 + 255) & ~(size_t)255);
static constexpr size_t OFF_W1   = OFF_TBF + (size_t)T_TOK*HDIM*2;        // 67MB
static constexpr size_t OFF_W2   = OFF_W1  + (size_t)NEXP*I2*HDIM*2;      // 33.5MB
static constexpr size_t OFF_ACT  = OFF_W2  + (size_t)NEXP*HDIM*IDIM*2;    // 16MB

union ABFrag { v16bf v; v8bf h[2]; };

__global__ __launch_bounds__(32) void zero_counts_kernel(int* counts) {
    if (threadIdx.x < NEXP) counts[threadIdx.x] = 0;
}

// fp32 -> bf16, 8 elements per thread
__global__ __launch_bounds__(256) void cvt_bf16_kernel(const float* __restrict__ src,
                                                       bf16* __restrict__ dst, int n8) {
    int i = blockIdx.x * 256 + threadIdx.x;
    if (i >= n8) return;
    const float4 a = ((const float4*)src)[2 * (size_t)i + 0];
    const float4 b = ((const float4*)src)[2 * (size_t)i + 1];
    v8bf o;
    o[0] = (bf16)a.x; o[1] = (bf16)a.y; o[2] = (bf16)a.z; o[3] = (bf16)a.w;
    o[4] = (bf16)b.x; o[5] = (bf16)b.y; o[6] = (bf16)b.z; o[7] = (bf16)b.w;
    *(v8bf*)(dst + (size_t)i * 8) = o;
}

__global__ __launch_bounds__(256) void copy_f4_kernel(const float4* __restrict__ src,
                                                      float4* __restrict__ dst, int n4) {
    int i = blockIdx.x * 256 + threadIdx.x;
    if (i < n4) dst[i] = src[i];
}

// one block per token: rmsnorm -> bf16 t, gate, top-4 softmax, bucket scatter
__global__ __launch_bounds__(256) void rms_gate_topk_kernel(
    const float* __restrict__ x, const float* __restrict__ nscale,
    const float* __restrict__ gw, const float* __restrict__ gb,
    bf16* __restrict__ tbf, int* __restrict__ counts,
    int* __restrict__ tokOf, float* __restrict__ wOf)
{
    const int t   = blockIdx.x;
    const int tid = threadIdx.x;
    const float* xr = x + (size_t)t * HDIM;

    __shared__ float red[256];
    float ss = 0.f;
    for (int j = tid; j < HDIM; j += 256) { float v = xr[j]; ss += v * v; }
    red[tid] = ss; __syncthreads();
    for (int s = 128; s > 0; s >>= 1) { if (tid < s) red[tid] += red[tid + s]; __syncthreads(); }
    __shared__ float sinv;
    if (tid == 0) sinv = rsqrtf(red[0] * (1.0f / HDIM) + 1e-5f);
    __syncthreads();
    const float inv = sinv;

    float acc[NEXP];
#pragma unroll
    for (int e = 0; e < NEXP; e++) acc[e] = 0.f;
    for (int j = tid; j < HDIM; j += 256) {
        float tv = xr[j] * inv * nscale[j];
        tbf[(size_t)t * HDIM + j] = (bf16)tv;
#pragma unroll
        for (int e = 0; e < NEXP; e++) acc[e] += tv * gw[e * HDIM + j];
    }
    __shared__ float gred[8][NEXP];
    const int lane = tid & 31, wv = tid >> 5;
#pragma unroll
    for (int e = 0; e < NEXP; e++) {
        float v = acc[e];
        for (int o = 16; o > 0; o >>= 1) v += __shfl_down(v, o, 32);
        if (lane == 0) gred[wv][e] = v;
    }
    __syncthreads();
    __shared__ float gfin[NEXP];
    if (tid < NEXP) {
        float v = gb[tid];
        for (int w = 0; w < 8; w++) v += gred[w][tid];
        gfin[tid] = v;
    }
    __syncthreads();
    if (tid == 0) {
        int idxs[TOPK]; float vals[TOPK]; unsigned used = 0;
        for (int k = 0; k < TOPK; k++) {
            float best = -3.402823466e38f; int bi = 0;
            for (int e = 0; e < NEXP; e++)
                if (!((used >> e) & 1u) && gfin[e] > best) { best = gfin[e]; bi = e; }
            used |= 1u << bi; idxs[k] = bi; vals[k] = best;
        }
        float mx = vals[0], sum = 0.f, ex[TOPK];
        for (int k = 0; k < TOPK; k++) { ex[k] = __expf(vals[k] - mx); sum += ex[k]; }
        float rs = 1.0f / sum;
        for (int k = 0; k < TOPK; k++) {
            int e   = idxs[k];
            int pos = atomicAdd(&counts[e], 1);
            tokOf[e * T_TOK + pos] = t;
            wOf [e * T_TOK + pos] = ex[k] * rs;
        }
    }
}

__global__ __launch_bounds__(32) void scan_kernel(const int* __restrict__ counts,
                                                  int* __restrict__ baseo) {
    if (threadIdx.x == 0) {
        int s = 0;
        for (int e = 0; e < NEXP; e++) { baseo[e] = s; s += counts[e]; }
    }
}

// grouped GEMM1 + SwiGLU: h = t @ W1[e]^T + b1 ; act = swiglu(h)
// grid: (I2/256, T/32, E), block 256 (8 waves; per wave: 2 M-tiles x 2 N-tiles)
__global__ __launch_bounds__(256) void gemm1_kernel(
    const bf16* __restrict__ tbf, const bf16* __restrict__ w1b,
    const float* __restrict__ b1, const int* __restrict__ counts,
    const int* __restrict__ baseo, const int* __restrict__ tokOf,
    bf16* __restrict__ act)
{
    const int e   = blockIdx.z;
    const int cnt = counts[e];
    const int m0  = blockIdx.y * 32;
    if (m0 >= cnt) return;
    const int rows = (cnt - m0 < 32) ? (cnt - m0) : 32;

    __shared__ bf16 As[32][AS_STR];   // 66KB: legal on gfx1250 (320KB/WGP)
    __shared__ int  stok[32];
    const int tid = threadIdx.x;
    if (tid < 32) {
        int r = tid < rows ? tid : rows - 1;
        stok[tid] = tokOf[e * T_TOK + m0 + r];
    }
    __syncthreads();
    for (int c = tid; c < 32 * 128; c += 256) {
        int r = c >> 7, ch = c & 127;
        uint4 v = *(const uint4*)(tbf + (size_t)stok[r] * HDIM + ch * 8);
        *(uint4*)(&As[r][ch * 8]) = v;
    }
    __syncthreads();

    const int wave = tid >> 5, lane = tid & 31;
    const int half = lane >> 4, nl = lane & 15;
    const int n0   = blockIdx.x * 256 + wave * 32;     // this wave: cols n0..n0+31
    const int off0 = half * 8;

    const bf16* pB0 = w1b + ((size_t)e * I2 + (n0 + nl))      * HDIM + off0;
    const bf16* pB1 = w1b + ((size_t)e * I2 + (n0 + 16 + nl)) * HDIM + off0;
    const bf16* aRow0 = &As[nl][off0];
    const bf16* aRow1 = &As[nl + 16][off0];

    v8f acc00 = {}; v8f acc01 = {}; v8f acc10 = {}; v8f acc11 = {};
    for (int kb = 0; kb < HDIM; kb += 32) {
        ABFrag a0, a1, b0, b1f;
        a0.h[0]  = *(const v8bf*)(aRow0 + kb);
        a0.h[1]  = *(const v8bf*)(aRow0 + kb + 16);
        a1.h[0]  = *(const v8bf*)(aRow1 + kb);
        a1.h[1]  = *(const v8bf*)(aRow1 + kb + 16);
        b0.h[0]  = *(const v8bf*)(pB0 + kb);
        b0.h[1]  = *(const v8bf*)(pB0 + kb + 16);
        b1f.h[0] = *(const v8bf*)(pB1 + kb);
        b1f.h[1] = *(const v8bf*)(pB1 + kb + 16);
        acc00 = __builtin_amdgcn_wmma_f32_16x16x32_bf16(false, a0.v, false, b0.v,
                                                        (short)0, acc00, false, false);
        acc01 = __builtin_amdgcn_wmma_f32_16x16x32_bf16(false, a0.v, false, b1f.v,
                                                        (short)0, acc01, false, false);
        acc10 = __builtin_amdgcn_wmma_f32_16x16x32_bf16(false, a1.v, false, b0.v,
                                                        (short)0, acc10, false, false);
        acc11 = __builtin_amdgcn_wmma_f32_16x16x32_bf16(false, a1.v, false, b1f.v,
                                                        (short)0, acc11, false, false);
    }

    const int slot0 = baseo[e] + m0;
    v8f accArr[2][2] = {{acc00, acc01}, {acc10, acc11}};
#pragma unroll
    for (int mt = 0; mt < 2; mt++) {
#pragma unroll
        for (int tIdx = 0; tIdx < 2; tIdx++) {
            const int n = n0 + 16 * tIdx + nl;
            const float bias = b1[(size_t)e * I2 + n];
#pragma unroll
            for (int r = 0; r < 8; r++) {
                const int m = 16 * mt + 8 * half + r;
                float h  = accArr[mt][tIdx][r] + bias;
                float hp = __shfl_xor(h, 1, 32);   // partner channel (odd<->even)
                float glu = fminf(h, LIMIT);
                float lin = fminf(fmaxf(hp, -LIMIT), LIMIT);
                float av  = glu * (1.0f / (1.0f + __expf(-ALPHA * glu))) * (lin + 1.0f);
                if (m < rows && ((n & 1) == 0))
                    act[(size_t)(slot0 + m) * IDIM + (n >> 1)] = (bf16)av;
            }
        }
    }
}

// grouped GEMM2: y = act @ W2[e]^T + b2 ; out += gate_w * y (f32 atomics)
// A-panel staged by the Tensor Data Mover (TDM): one 2D tile 32 x 1024 elem (bf16),
// with TDM LDS padding (16B per 1024B) reproducing the bank-spread layout.
// LDS layout: 64 half-rows of HR_B=1040 bytes (1024B data + 16B pad).
// grid: (H/256, T/32, E), block 256
__global__ __launch_bounds__(256) void gemm2_kernel(
    const bf16* __restrict__ act, const bf16* __restrict__ w2b,
    const float* __restrict__ b2, const int* __restrict__ counts,
    const int* __restrict__ baseo, const int* __restrict__ tokOf,
    const float* __restrict__ wOf, float* __restrict__ out)
{
    const int e   = blockIdx.z;
    const int cnt = counts[e];
    const int m0  = blockIdx.y * 32;
    if (m0 >= cnt) return;
    const int rows = (cnt - m0 < 32) ? (cnt - m0) : 32;

    __shared__ __attribute__((aligned(16))) char As2[32 * 2 * HR_B];   // 66,560B
    __shared__ int   stok[32];
    __shared__ float swgt[32];
    const int tid  = threadIdx.x;
    const int wave = tid >> 5, lane = tid & 31;
    const int slot0 = baseo[e] + m0;
    if (tid < 32) {
        int r = tid < rows ? tid : rows - 1;
        stok[tid] = tokOf[e * T_TOK + m0 + r];
        swgt[tid] = wOf [e * T_TOK + m0 + r];
    }

#if HAS_TDM
    if (wave == 0) {
        // ---- Tensor DMA descriptor (ISA 8.3/8.4) ----
        unsigned ldsAddr = (unsigned)(size_t)(&As2[0]);
        unsigned long long ga =
            (unsigned long long)(size_t)(act + (size_t)slot0 * IDIM);
        u32x4 g0;
        g0[0] = 1u;                                   // count=1 valid, user mode
        g0[1] = ldsAddr;                              // lds_addr
        g0[2] = (unsigned)ga;                         // global_addr[31:0]
        g0[3] = (unsigned)((ga >> 32) & 0x01FFFFFFu)  // global_addr[56:32]
              | (2u << 30);                           // type = 2 (image)
        i32x8 g1;
        g1[0] = (int)((1u << 16)      // data_size = 1 -> 2 bytes
                    | (1u << 20)      // pad_enable
                    | (7u << 22)      // pad_interval: 256 DWORDs = 1024B
                    | (3u << 25));    // pad_amount:   4 DWORDs  = 16B
        g1[1] = (int)(1024u << 16);   // tensor_dim0 = 1024 elems
        g1[2] = (int)(8192u << 16);   // tensor_dim1 = 8192 rows (all act slots)
        g1[3] = (int)(1024u << 16);   // tile_dim0 = 1024
        g1[4] = 32;                   // tile_dim1 = 32 rows, tile_dim2 = 0
        g1[5] = 1024;                 // tensor_dim0_stride = 1024 elems
        g1[6] = 0;
        g1[7] = 0;
        i32x4 z4 = {0, 0, 0, 0};              // groups 2/3: dims 2..4 unused
        i32x8 z8 = {0, 0, 0, 0, 0, 0, 0, 0};  // 6-arg toolchain variant: extra group
        __builtin_amdgcn_tensor_load_to_lds(g0, g1, z4, z4, z8, 0);
#if __has_builtin(__builtin_amdgcn_s_wait_tensorcnt)
        __builtin_amdgcn_s_wait_tensorcnt(0);
#else
        asm volatile("s_wait_tensorcnt 0x0" ::: "memory");
#endif
    }
    __syncthreads();
#else
    __syncthreads();
    for (int c = tid; c < 32 * 128; c += 256) {
        int r = c >> 7, ch = c & 127;
        int rr = r < rows ? r : rows - 1;
        int s  = ch * 8;                                // element index in row
        uint4 v = *(const uint4*)(act + (size_t)(slot0 + rr) * IDIM + s);
        *(uint4*)(&As2[(2 * r + (s >> 9)) * HR_B + (s & 511) * 2]) = v;
    }
    __syncthreads();
#endif

    const int half = lane >> 4, nl = lane & 15;
    const int n0   = blockIdx.x * 256 + wave * 32;
    const int off0 = half * 8;

    const bf16* pB0 = w2b + ((size_t)e * HDIM + (n0 + nl))      * IDIM + off0;
    const bf16* pB1 = w2b + ((size_t)e * HDIM + (n0 + 16 + nl)) * IDIM + off0;
    const char* aBase0 = As2 + (2 * nl) * HR_B;
    const char* aBase1 = As2 + (2 * (nl + 16)) * HR_B;

    v8f acc00 = {}; v8f acc01 = {}; v8f acc10 = {}; v8f acc11 = {};
    for (int kb = 0; kb < IDIM; kb += 32) {
        const int s0 = kb + off0, s1 = kb + off0 + 16;
        ABFrag a0, a1, b0, b1f;
        a0.h[0]  = *(const v8bf*)(aBase0 + (s0 >> 9) * HR_B + (s0 & 511) * 2);
        a0.h[1]  = *(const v8bf*)(aBase0 + (s1 >> 9) * HR_B + (s1 & 511) * 2);
        a1.h[0]  = *(const v8bf*)(aBase1 + (s0 >> 9) * HR_B + (s0 & 511) * 2);
        a1.h[1]  = *(const v8bf*)(aBase1 + (s1 >> 9) * HR_B + (s1 & 511) * 2);
        b0.h[0]  = *(const v8bf*)(pB0 + kb);
        b0.h[1]  = *(const v8bf*)(pB0 + kb + 16);
        b1f.h[0] = *(const v8bf*)(pB1 + kb);
        b1f.h[1] = *(const v8bf*)(pB1 + kb + 16);
        acc00 = __builtin_amdgcn_wmma_f32_16x16x32_bf16(false, a0.v, false, b0.v,
                                                        (short)0, acc00, false, false);
        acc01 = __builtin_amdgcn_wmma_f32_16x16x32_bf16(false, a0.v, false, b1f.v,
                                                        (short)0, acc01, false, false);
        acc10 = __builtin_amdgcn_wmma_f32_16x16x32_bf16(false, a1.v, false, b0.v,
                                                        (short)0, acc10, false, false);
        acc11 = __builtin_amdgcn_wmma_f32_16x16x32_bf16(false, a1.v, false, b1f.v,
                                                        (short)0, acc11, false, false);
    }

    v8f accArr[2][2] = {{acc00, acc01}, {acc10, acc11}};
#pragma unroll
    for (int mt = 0; mt < 2; mt++) {
#pragma unroll
        for (int tIdx = 0; tIdx < 2; tIdx++) {
            const int n = n0 + 16 * tIdx + nl;
            const float bias = b2[(size_t)e * HDIM + n];
#pragma unroll
            for (int r = 0; r < 8; r++) {
                const int m = 16 * mt + 8 * half + r;
                if (m < rows) {
                    float y = accArr[mt][tIdx][r] + bias;
                    atomicAdd(&out[(size_t)stok[m] * HDIM + n], swgt[m] * y);
                }
            }
        }
    }
}

extern "C" void kernel_launch(void* const* d_in, const int* in_sizes, int n_in,
                              void* d_out, int out_size, void* d_ws, size_t ws_size,
                              hipStream_t stream) {
    (void)in_sizes; (void)n_in; (void)out_size; (void)ws_size;
    const float* x   = (const float*)d_in[0];
    const float* nsc = (const float*)d_in[1];
    const float* gw  = (const float*)d_in[2];
    const float* gb  = (const float*)d_in[3];
    const float* w1  = (const float*)d_in[4];
    const float* b1  = (const float*)d_in[5];
    const float* w2  = (const float*)d_in[6];
    const float* b2  = (const float*)d_in[7];
    float* out = (float*)d_out;

    char* ws = (char*)d_ws;
    int*   counts = (int*)  (ws + OFF_CNT);
    int*   baseo  = (int*)  (ws + OFF_BASE);
    int*   tokOf  = (int*)  (ws + OFF_TOK);
    float* wOf    = (float*)(ws + OFF_WGT);
    bf16*  tbf    = (bf16*) (ws + OFF_TBF);
    bf16*  w1b    = (bf16*) (ws + OFF_W1);
    bf16*  w2b    = (bf16*) (ws + OFF_W2);
    bf16*  act    = (bf16*) (ws + OFF_ACT);

    zero_counts_kernel<<<1, 32, 0, stream>>>(counts);

    const int n8_w1 = NEXP * I2 * HDIM / 8;     // 4,194,304
    const int n8_w2 = NEXP * HDIM * IDIM / 8;   // 2,097,152
    cvt_bf16_kernel<<<(n8_w1 + 255) / 256, 256, 0, stream>>>(w1, w1b, n8_w1);
    cvt_bf16_kernel<<<(n8_w2 + 255) / 256, 256, 0, stream>>>(w2, w2b, n8_w2);

    const int n4 = T_TOK * HDIM / 4;
    copy_f4_kernel<<<(n4 + 255) / 256, 256, 0, stream>>>((const float4*)x, (float4*)out, n4);

    rms_gate_topk_kernel<<<T_TOK, 256, 0, stream>>>(x, nsc, gw, gb, tbf, counts, tokOf, wOf);
    scan_kernel<<<1, 32, 0, stream>>>(counts, baseo);

    gemm1_kernel<<<dim3(I2 / 256, T_TOK / 32, NEXP), 256, 0, stream>>>(
        tbf, w1b, b1, counts, baseo, tokOf, act);
    gemm2_kernel<<<dim3(HDIM / 256, T_TOK / 32, NEXP), 256, 0, stream>>>(
        act, w2b, b2, counts, baseo, tokOf, wOf, out);
}
